// WorldModel_42494406427276
// MI455X (gfx1250) — matrix operand order repaired and checked by reading, not compiled, not run on hardware
//
#include <hip/hip_runtime.h>

typedef __attribute__((ext_vector_type(16))) _Float16 v16h;
typedef __attribute__((ext_vector_type(8)))  _Float16 v8h;
typedef __attribute__((ext_vector_type(8)))  float    v8f;

#define WMMA_F16(a, b, c) \
    __builtin_amdgcn_wmma_f32_16x16x32_f16(false, (a), false, (b), (short)0, (c), false, false)

// build v16h from two aligned 16-byte LDS vectors (2x ds_load_b128)
__device__ __forceinline__ v16h cat16(const _Float16* lo, const _Float16* hi) {
    v8h a = *(const v8h*)lo;
    v8h b = *(const v8h*)hi;
    return __builtin_shufflevector(a, b, 0, 1, 2, 3, 4, 5, 6, 7, 8, 9, 10, 11, 12, 13, 14, 15);
}

// ---------------------------------------------------------------------------
// 1) conv1 (2->16, 3x3, SAME, relu) + conv2 (16->16, 3x3, SAME, relu)
// ---------------------------------------------------------------------------
__global__ void conv_kernel(const float* __restrict__ obs,
                            const float* __restrict__ c1w, const float* __restrict__ c1b,
                            const float* __restrict__ c2w, const float* __restrict__ c2b,
                            float* __restrict__ cell) {
    __shared__ float h1[64][16];
    int t = threadIdx.x;            // 0..63
    int r = t >> 3, c = t & 7;
    for (int oc = 0; oc < 16; ++oc) {
        float s = c1b[oc];
        for (int ic = 0; ic < 2; ++ic)
            for (int kr = 0; kr < 3; ++kr)
                for (int kc = 0; kc < 3; ++kc) {
                    int rr = r + kr - 1, cc = c + kc - 1;
                    if (rr >= 0 && rr < 8 && cc >= 0 && cc < 8)
                        s += obs[ic * 64 + rr * 8 + cc] * c1w[((oc * 2 + ic) * 3 + kr) * 3 + kc];
                }
        h1[t][oc] = fmaxf(s, 0.f);
    }
    __syncthreads();
    for (int oc = 0; oc < 16; ++oc) {
        float s = c2b[oc];
        for (int ic = 0; ic < 16; ++ic)
            for (int kr = 0; kr < 3; ++kr)
                for (int kc = 0; kc < 3; ++kc) {
                    int rr = r + kr - 1, cc = c + kc - 1;
                    if (rr >= 0 && rr < 8 && cc >= 0 && cc < 8)
                        s += h1[rr * 8 + cc][ic] * c2w[((oc * 16 + ic) * 3 + kr) * 3 + kc];
                }
        cell[t * 16 + oc] = fmaxf(s, 0.f);
    }
}

// ---------------------------------------------------------------------------
// 2) pairwise token build: seq[t=i*64+j] = [cell_i(16), cell_j(16), dx, dy]
// ---------------------------------------------------------------------------
__global__ void token_kernel(const float* __restrict__ cell, float* __restrict__ x) {
    int t = blockIdx.x * blockDim.x + threadIdx.x;
    if (t >= 4096) return;
    int i = t >> 6, j = t & 63;
    float* o = x + (size_t)t * 34;
    for (int d = 0; d < 16; ++d) o[d]      = cell[i * 16 + d];
    for (int d = 0; d < 16; ++d) o[16 + d] = cell[j * 16 + d];
    float ri = (float)(i >> 3), ci = (float)(i & 7);
    float rj = (float)(j >> 3), cj = (float)(j & 7);
    o[32] = (ri - rj) / 3.5f;
    o[33] = (ci - cj) / 3.5f;
}

// ---------------------------------------------------------------------------
// 3) Generic WMMA GEMM: C[M,N] = act(A[M,K] * W[N,K]^T + bias)
//    One wave per block computes a 16x16 tile; K-loop in steps of 32.
//    M is a multiple of 16 (4096). Full-K chunks take an unguarded fast path.
// ---------------------------------------------------------------------------
template <bool A_HALF, bool RELU, bool OUT_HALF>
__global__ void wmma_gemm(const void* __restrict__ Aptr, const float* __restrict__ W,
                          const float* __restrict__ bias, void* __restrict__ Cptr,
                          int M, int N, int K) {
    __shared__ __align__(16) _Float16 As[16][32];
    __shared__ __align__(16) _Float16 Bs[32][16];
    const float*    Af = (const float*)Aptr;
    const _Float16* Ah = (const _Float16*)Aptr;
    int lane = threadIdx.x;                 // block = 32 threads = 1 wave
    int m0 = blockIdx.x * 16, n0 = blockIdx.y * 16;
    int l16 = lane & 15, sub = lane >> 4;
    // per-lane half-row for the A fill: row ar, k offsets ak..ak+15
    int ar = lane >> 1, ak = (lane & 1) * 16;
    v8f acc = {};
    for (int k0 = 0; k0 < K; k0 += 32) {
        if (k0 + 32 <= K) {                 // full chunk: no k guards
            const size_t abase = (size_t)(m0 + ar) * K + k0 + ak;
            for (int e = 0; e < 16; ++e) {
                float v = A_HALF ? (float)Ah[abase + e] : Af[abase + e];
                As[ar][ak + e] = (_Float16)v;
            }
            for (int e = 0; e < 16; ++e) {
                int idx = lane * 16 + e;
                int kk = idx >> 4, n = idx & 15;
                int col = n0 + n;
                float v = (col < N) ? W[(size_t)col * K + k0 + kk] : 0.f;
                Bs[kk][n] = (_Float16)v;
            }
        } else {                            // tail chunk: zero-pad K
            for (int e = 0; e < 16; ++e) {
                int k = k0 + ak + e;
                float v = 0.f;
                if (k < K) v = A_HALF ? (float)Ah[(size_t)(m0 + ar) * K + k]
                                      : Af[(size_t)(m0 + ar) * K + k];
                As[ar][ak + e] = (_Float16)v;
            }
            for (int e = 0; e < 16; ++e) {
                int idx = lane * 16 + e;
                int kk = idx >> 4, n = idx & 15;
                int col = n0 + n, k = k0 + kk;
                float v = (col < N && k < K) ? W[(size_t)col * K + k] : 0.f;
                Bs[kk][n] = (_Float16)v;
            }
        }
        __syncthreads();
        v16h a = cat16(&As[l16][sub * 8], &As[l16][16 + sub * 8]);
        v16h b; for (int s = 0; s < 16; ++s) b[s] = Bs[(s < 8 ? 8 * sub + s : 8 + 8 * sub + s)][l16];
        acc = WMMA_F16(a, b, acc);
        __syncthreads();
    }
    for (int r = 0; r < 8; ++r) {
        int row = m0 + r + 8 * sub, col = n0 + l16;
        if (row < M && col < N) {
            float v = acc[r] + bias[col];
            if (RELU) v = fmaxf(v, 0.f);
            if (OUT_HALF) ((_Float16*)Cptr)[(size_t)row * N + col] = (_Float16)v;
            else          ((float*)Cptr)[(size_t)row * N + col]    = v;
        }
    }
}

// ---------------------------------------------------------------------------
// 4) Flash attention: S=4096, hd=17 (padded to 32), NH=2.
//    Block = 128 threads (4 waves); each wave owns 16 query rows; block covers
//    64 rows of one head. 32-key tiles: 2 score WMMAs + online max +
//    2 PV WMMAs. The softmax denominator is computed BY the matrix unit:
//    V's padded dim 17 is set to 1.0, so o1 column 17 accumulates rowsum(P)
//    with the correct running rescale -- no per-iteration sum butterfly.
// ---------------------------------------------------------------------------
__global__ void flash_attn_kernel(const float* __restrict__ qkv, float* __restrict__ out) {
    __shared__ __align__(16) _Float16 Qs[64][32];
    __shared__ __align__(16) _Float16 KsT[32][32];   // [key][dim]
    __shared__ __align__(16) _Float16 Vs[32][32];    // [dim][key], dim17 = 1.0
    __shared__ __align__(16) _Float16 Ps[4][16][32];
    const int S = 4096, LDQ = 102;
    int h  = blockIdx.y;
    int q0 = blockIdx.x * 64;
    int tid = threadIdx.x;
    int w = tid >> 5, lane = tid & 31, l16 = lane & 15, sub = lane >> 4;
    const float scale = 0.24253562503633297f;   // 1/sqrt(17)

    for (int idx = tid; idx < 64 * 32; idx += 128) {
        int r = idx >> 5, kk = idx & 31;
        float v = (kk < 17) ? qkv[(size_t)(q0 + r) * LDQ + h * 17 + kk] * scale : 0.f;
        Qs[r][kk] = (_Float16)v;
    }
    __syncthreads();

    v16h qa = cat16(&Qs[w * 16 + l16][sub * 8], &Qs[w * 16 + l16][16 + sub * 8]);

    float m[8];
    v8f o0 = {}, o1 = {};
    for (int r = 0; r < 8; ++r) m[r] = -1e30f;

    for (int j0 = 0; j0 < S; j0 += 32) {
        __syncthreads();
        for (int idx = tid; idx < 32 * 32; idx += 128) {
            int r = idx >> 5, kk = idx & 31;   // r = key within tile, kk = head dim
            float kv = (kk < 17) ? qkv[(size_t)(j0 + r) * LDQ + 34 + h * 17 + kk] : 0.f;
            float vv = (kk < 17) ? qkv[(size_t)(j0 + r) * LDQ + 68 + h * 17 + kk]
                                 : (kk == 17 ? 1.0f : 0.f);
            KsT[r][kk] = (_Float16)kv;         // [key][dim] store
            Vs[kk][r]  = (_Float16)vv;
        }
        __syncthreads();

        v16h b0  = cat16(&KsT[l16][sub * 8],      &KsT[l16][16 + sub * 8]);
        v16h b1  = cat16(&KsT[16 + l16][sub * 8], &KsT[16 + l16][16 + sub * 8]);
        v16h vb0 = cat16(&Vs[l16][sub * 8],       &Vs[l16][16 + sub * 8]);
        v16h vb1 = cat16(&Vs[16 + l16][sub * 8],  &Vs[16 + l16][16 + sub * 8]);

        v8f s0 = {}, s1 = {};
        s0 = WMMA_F16(qa, b0, s0);
        s1 = WMMA_F16(qa, b1, s1);

        for (int r = 0; r < 8; ++r) {
            float mx = fmaxf(s0[r], s1[r]);
            for (int off = 1; off < 16; off <<= 1) mx = fmaxf(mx, __shfl_xor(mx, off, 32));
            float mn   = fmaxf(m[r], mx);
            float corr = __expf(m[r] - mn);
            m[r] = mn;
            o0[r] *= corr; o1[r] *= corr;
            Ps[w][r + 8 * sub][l16]      = (_Float16)__expf(s0[r] - mn);
            Ps[w][r + 8 * sub][16 + l16] = (_Float16)__expf(s1[r] - mn);
        }
        __builtin_amdgcn_wave_barrier();
        v16h pa = cat16(&Ps[w][l16][sub * 8], &Ps[w][l16][16 + sub * 8]);
        o0 = WMMA_F16(pa, vb0, o0);
        o1 = WMMA_F16(pa, vb1, o1);
    }

    for (int r = 0; r < 8; ++r) {
        // softmax denominator = o1 column (dim) 17, held by lane l16==1 of this half
        float denom = __shfl(o1[r], 16 * sub + 1, 32);
        float inv = 1.f / denom;
        int row = q0 + w * 16 + r + 8 * sub;
        int d0 = l16;                   // always < 17
        out[(size_t)row * 34 + h * 17 + d0] = o0[r] * inv;
        int d1 = 16 + l16;              // only d1 == 16 is a real dim
        if (d1 < 17) out[(size_t)row * 34 + h * 17 + d1] = o1[r] * inv;
    }
}

// ---------------------------------------------------------------------------
// 5) x = LayerNorm(x + res) over D=34, one row per thread
// ---------------------------------------------------------------------------
__global__ void add_ln_kernel(float* __restrict__ x, const float* __restrict__ res,
                              const float* __restrict__ g, const float* __restrict__ b) {
    int row = blockIdx.x * blockDim.x + threadIdx.x;
    if (row >= 4096) return;
    float v[34];
    float mean = 0.f;
    for (int d = 0; d < 34; ++d) {
        v[d] = x[(size_t)row * 34 + d] + res[(size_t)row * 34 + d];
        mean += v[d];
    }
    mean *= (1.f / 34.f);
    float var = 0.f;
    for (int d = 0; d < 34; ++d) { float t = v[d] - mean; var += t * t; }
    var *= (1.f / 34.f);
    float inv = rsqrtf(var + 1e-5f);
    for (int d = 0; d < 34; ++d)
        x[(size_t)row * 34 + d] = (v[d] - mean) * inv * g[d] + b[d];
}

// ---------------------------------------------------------------------------
// 6) argmax(obs[0]) -> gather 64 rows -> mean -> action embed -> head (16)
// ---------------------------------------------------------------------------
__global__ void head_kernel(const float* __restrict__ obs, const float* __restrict__ x,
                            const float* __restrict__ act_w, const float* __restrict__ act_b,
                            const float* __restrict__ head_w, const float* __restrict__ head_b,
                            const int* __restrict__ action, float* __restrict__ out) {
    __shared__ float hin[42];
    __shared__ int agent;
    int t = threadIdx.x;   // 64
    if (t == 0) {
        int best = 0; float bv = obs[0];
        for (int i = 1; i < 64; ++i)
            if (obs[i] > bv) { bv = obs[i]; best = i; }
        agent = best;
    }
    __syncthreads();
    if (t < 34) {
        float s = 0.f;
        for (int j = 0; j < 64; ++j)
            s += x[(size_t)(agent * 64 + j) * 34 + t];
        hin[t] = s * (1.f / 64.f);
    } else if (t < 42) {
        int o = t - 34;
        hin[t] = fmaxf(act_w[o * 4 + *action] + act_b[o], 0.f);
    }
    __syncthreads();
    if (t < 16) {
        float s = head_b[t];
        for (int d = 0; d < 42; ++d) s += head_w[t * 42 + d] * hin[d];
        out[t] = s;
    }
}

// ---------------------------------------------------------------------------
// host-side orchestration
// ---------------------------------------------------------------------------
extern "C" void kernel_launch(void* const* d_in, const int* in_sizes, int n_in,
                              void* d_out, int out_size, void* d_ws, size_t ws_size,
                              hipStream_t stream) {
    (void)in_sizes; (void)n_in; (void)out_size; (void)ws_size;
    const float* obs    = (const float*)d_in[0];
    const int*   action = (const int*)d_in[1];
    const float* c1w = (const float*)d_in[2]; const float* c1b = (const float*)d_in[3];
    const float* c2w = (const float*)d_in[4]; const float* c2b = (const float*)d_in[5];
    const float* act_w  = (const float*)d_in[30]; const float* act_b  = (const float*)d_in[31];
    const float* head_w = (const float*)d_in[32]; const float* head_b = (const float*)d_in[33];

    // workspace layout (bytes)
    char* ws = (char*)d_ws;
    float*    cell = (float*)(ws + 0);                 //  4096 B
    float*    x    = (float*)(ws + 4096);              //  4096*34*4   = 557056
    float*    qkv  = (float*)(ws + 561152);            //  4096*102*4  = 1671168
    float*    attn = (float*)(ws + 2232320);           //  557056
    float*    tmp  = (float*)(ws + 2789376);           //  557056
    _Float16* hbuf = (_Float16*)(ws + 3346432);        //  4096*2048*2 = 16777216

    conv_kernel<<<1, 64, 0, stream>>>(obs, c1w, c1b, c2w, c2b, cell);
    token_kernel<<<32, 128, 0, stream>>>(cell, x);

    for (int L = 0; L < 2; ++L) {
        const float* in_w  = (const float*)d_in[6  + 12 * L];
        const float* in_b  = (const float*)d_in[7  + 12 * L];
        const float* out_w = (const float*)d_in[8  + 12 * L];
        const float* out_b = (const float*)d_in[9  + 12 * L];
        const float* ff1_w = (const float*)d_in[10 + 12 * L];
        const float* ff1_b = (const float*)d_in[11 + 12 * L];
        const float* ff2_w = (const float*)d_in[12 + 12 * L];
        const float* ff2_b = (const float*)d_in[13 + 12 * L];
        const float* ln1_g = (const float*)d_in[14 + 12 * L];
        const float* ln1_b = (const float*)d_in[15 + 12 * L];
        const float* ln2_g = (const float*)d_in[16 + 12 * L];
        const float* ln2_b = (const float*)d_in[17 + 12 * L];

        // QKV: [4096,34] x [34,102]
        wmma_gemm<false, false, false><<<dim3(256, 7), 32, 0, stream>>>(x, in_w, in_b, qkv, 4096, 102, 34);
        // flash attention -> attn [4096,34]
        flash_attn_kernel<<<dim3(64, 2), 128, 0, stream>>>(qkv, attn);
        // out-proj: [4096,34] x [34,34]
        wmma_gemm<false, false, false><<<dim3(256, 3), 32, 0, stream>>>(attn, out_w, out_b, tmp, 4096, 34, 34);
        add_ln_kernel<<<16, 256, 0, stream>>>(x, tmp, ln1_g, ln1_b);
        // FF1: relu([4096,34] x [34,2048]) -> f16
        wmma_gemm<false, true, true><<<dim3(256, 128), 32, 0, stream>>>(x, ff1_w, ff1_b, hbuf, 4096, 2048, 34);
        // FF2: [4096,2048](f16) x [2048,34]
        wmma_gemm<true, false, false><<<dim3(256, 3), 32, 0, stream>>>(hbuf, ff2_w, ff2_b, tmp, 4096, 34, 2048);
        add_ln_kernel<<<16, 256, 0, stream>>>(x, tmp, ln2_g, ln2_b);
    }

    head_kernel<<<1, 64, 0, stream>>>(obs, x, act_w, act_b, head_w, head_b, action, (float*)d_out);
}
